// VpMoETPAAttention_47871705481251
// MI455X (gfx1250) — matrix operand order repaired and checked by reading, not compiled, not run on hardware
//
#include <hip/hip_runtime.h>
#include <hip/hip_bf16.h>
#include <math.h>

// Problem constants (match reference)
#define B_   2
#define S_   2048
#define D_   2048
#define H_   16
#define KVH_ 4
#define HD_  128
#define QR_  6
#define R_   2
#define NEG_ (-1000000000.0f)
#define SCALING_ 0.08838834764831845f   // 128^-0.5

typedef __attribute__((ext_vector_type(16))) __bf16        v16bf;
typedef __attribute__((ext_vector_type(8)))  float         v8f;
typedef __attribute__((ext_vector_type(4)))  unsigned int  v4u;
typedef __attribute__((ext_vector_type(4)))  unsigned int  u32x4;
typedef __attribute__((ext_vector_type(8)))  int           i32x8;
typedef __attribute__((ext_vector_type(4)))  int           i32x4;

#if defined(__has_builtin)
# if __has_builtin(__builtin_amdgcn_tensor_load_to_lds)
#  define HAVE_TDM 1
# endif
#endif
#ifndef HAVE_TDM
# define HAVE_TDM 0
#endif

union BF16Frag { v16bf v; unsigned short u[16]; v4u q[2]; };

__device__ __forceinline__ unsigned short f2bf(float f) {
  unsigned int u = __float_as_uint(f);
  unsigned int r = u + 0x7FFFu + ((u >> 16) & 1u);   // round-to-nearest-even
  return (unsigned short)(r >> 16);
}

__device__ __forceinline__ void wait_tensorcnt0() {
#if defined(__has_builtin) && __has_builtin(__builtin_amdgcn_s_wait_tensorcnt)
  __builtin_amdgcn_s_wait_tensorcnt(0);
#else
  asm volatile("s_wait_tensorcnt 0" ::: "memory");
#endif
}

#if HAVE_TDM
// 2D TDM load: tile (tile_d0 x tile_d1) of 2-byte elements, row stride stride0
// elements, from global `gp` into LDS offset `lds_byte_off`.
// D# packing per CDNA5 ISA 8.3/8.4 (group0: count/lds/global/type; group1:
// data_size, tensor dims, tile dims, stride).
__device__ __forceinline__ void tdm_load_2d(unsigned int lds_byte_off,
                                            const void* gp,
                                            unsigned int tile_d0,
                                            unsigned int tile_d1,
                                            unsigned int stride0) {
  unsigned long long ga = (unsigned long long)(uintptr_t)gp;
  u32x4 g0;
  g0[0] = 1u;                                        // count=1, user mode
  g0[1] = lds_byte_off;                              // lds_addr
  g0[2] = (unsigned int)(ga & 0xFFFFFFFFu);          // global_addr[31:0]
  g0[3] = (unsigned int)((ga >> 32) & 0x01FFFFFFu)   // global_addr[56:32]
          | 0x80000000u;                             // type=2 ("image")
  i32x8 g1;
  g1[0] = 0x00010000;                                // data_size=1 -> 2 bytes
  g1[1] = (int)((tile_d0 & 0xFFFFu) << 16);          // tensor_dim0[15:0] @ bits63:48
  g1[2] = (int)(((tile_d0 >> 16) & 0xFFFFu) |        // tensor_dim0[31:16]
                ((tile_d1 & 0xFFFFu) << 16));        // tensor_dim1[15:0]
  g1[3] = (int)(((tile_d1 >> 16) & 0xFFFFu) |        // tensor_dim1[31:16]
                ((tile_d0 & 0xFFFFu) << 16));        // tile_dim0
  g1[4] = (int)(tile_d1 & 0xFFFFu);                  // tile_dim1 (tile_dim2=0)
  g1[5] = (int)stride0;                              // tensor_dim0_stride[31:0]
  g1[6] = 0;
  g1[7] = 0;
  i32x4 z4 = {0, 0, 0, 0};
#if defined(__clang_major__) && (__clang_major__ >= 23)
  i32x8 z8 = {0, 0, 0, 0, 0, 0, 0, 0};
  __builtin_amdgcn_tensor_load_to_lds(g0, g1, z4, z4, z8, 0);
#else
  __builtin_amdgcn_tensor_load_to_lds(g0, g1, z4, z4, 0);
#endif
}
#endif  // HAVE_TDM

// ---------------------------------------------------------------- converters
__global__ void cvt_f32_bf16(const float* __restrict__ src,
                             unsigned short* __restrict__ dst, int n) {
  int i = blockIdx.x * blockDim.x + threadIdx.x;
  if (i < n) dst[i] = f2bf(src[i]);
}

// src: K x N (row-major f32)  ->  dst: N x K (row-major bf16)
__global__ void transpose_cvt(const float* __restrict__ src,
                              unsigned short* __restrict__ dst, int K, int N) {
  int i = blockIdx.x * blockDim.x + threadIdx.x;
  if (i >= K * N) return;
  int n = i / K, k = i - n * K;
  dst[i] = f2bf(src[k * N + n]);
}

// ---------------------------------------------------------------- WMMA GEMM
// C(MxN,f32) = A(MxKd,bf16 row-major) * Bt(NxKd,bf16 row-major)^T
// One wave computes a 32x32 C tile (2x2 WMMA register block); 4 waves/block.
// Out-of-range B rows are clamped (their C columns are never stored), so the
// K-loop is branch-free and EXEC stays all-ones throughout (WMMA requirement).
__global__ __launch_bounds__(128) void gemm_bf16_wmma(
    const unsigned short* __restrict__ A,
    const unsigned short* __restrict__ Bt,
    float* __restrict__ C, int M, int N, int Kd) {
  const int lane   = threadIdx.x & 31;
  const int widx   = threadIdx.x >> 5;
  const int lane16 = lane & 15;
  const int half   = lane >> 4;
  const int mt32 = M >> 5;              // M multiple of 32
  const int nt32 = (N + 31) >> 5;
  int tile = blockIdx.x * 4 + widx;
  if (tile >= mt32 * nt32) return;      // wave-uniform exit
  const int mt = tile / nt32;
  const int nt = tile - mt * nt32;
  const int m0 = mt * 32, n0 = nt * 32;
  const unsigned short* ar0 = A + (size_t)(m0 + lane16) * Kd;
  const unsigned short* ar1 = A + (size_t)(m0 + 16 + lane16) * Kd;
  int bn0 = n0 + lane16;      bn0 = (bn0 < N) ? bn0 : (N - 1);
  int bn1 = n0 + 16 + lane16; bn1 = (bn1 < N) ? bn1 : (N - 1);
  const unsigned short* br0 = Bt + (size_t)bn0 * Kd;
  const unsigned short* br1 = Bt + (size_t)bn1 * Kd;
  v8f acc00 = {}, acc01 = {}, acc10 = {}, acc11 = {};
  for (int kk = 0; kk < Kd; kk += 32) {
    BF16Frag a0, a1, b0, b1;
    a0.q[0] = *(const v4u*)(ar0 + kk + 8 * half);
    a0.q[1] = *(const v4u*)(ar0 + kk + 16 + 8 * half);
    a1.q[0] = *(const v4u*)(ar1 + kk + 8 * half);
    a1.q[1] = *(const v4u*)(ar1 + kk + 16 + 8 * half);
    b0.q[0] = *(const v4u*)(br0 + kk + 8 * half);
    b0.q[1] = *(const v4u*)(br0 + kk + 16 + 8 * half);
    b1.q[0] = *(const v4u*)(br1 + kk + 8 * half);
    b1.q[1] = *(const v4u*)(br1 + kk + 16 + 8 * half);
    acc00 = __builtin_amdgcn_wmma_f32_16x16x32_bf16(false, a0.v, false, b0.v, (short)0, acc00, false, false);
    acc01 = __builtin_amdgcn_wmma_f32_16x16x32_bf16(false, a0.v, false, b1.v, (short)0, acc01, false, false);
    acc10 = __builtin_amdgcn_wmma_f32_16x16x32_bf16(false, a1.v, false, b0.v, (short)0, acc10, false, false);
    acc11 = __builtin_amdgcn_wmma_f32_16x16x32_bf16(false, a1.v, false, b1.v, (short)0, acc11, false, false);
  }
  const int c0 = n0 + lane16, c1 = n0 + 16 + lane16;
#pragma unroll
  for (int i = 0; i < 8; ++i) {
    int r0 = m0 + i + 8 * half;
    int r1 = r0 + 16;
    if (c0 < N) {
      C[(size_t)r0 * N + c0] = acc00[i];
      C[(size_t)r1 * N + c0] = acc10[i];
    }
    if (c1 < N) {
      C[(size_t)r0 * N + c1] = acc01[i];
      C[(size_t)r1 * N + c1] = acc11[i];
    }
  }
}

// ---------------------------------------------------------------- RoPE + rank combine
__global__ __launch_bounds__(128) void qkv_combine(
    const float* __restrict__ Aq, const float* __restrict__ Ak,
    const float* __restrict__ Av, const float* __restrict__ Bq,
    const float* __restrict__ Bk, const float* __restrict__ Bv,
    const float* __restrict__ cosb, const float* __restrict__ sinb,
    unsigned short* __restrict__ qo, unsigned short* __restrict__ ko,
    unsigned short* __restrict__ vto) {
  const int tok = blockIdx.x;          // b*S + s
  const int b = tok / S_;
  const int s = tok - b * S_;
  const int d = threadIdx.x;           // 0..127
  __shared__ float sAq[H_ * QR_];
  __shared__ float sAk[KVH_ * R_];
  __shared__ float sAv[KVH_ * R_];
  if (threadIdx.x < H_ * QR_) sAq[threadIdx.x] = Aq[(size_t)tok * (H_ * QR_) + threadIdx.x];
  if (threadIdx.x < KVH_ * R_) {
    sAk[threadIdx.x] = Ak[tok * (KVH_ * R_) + threadIdx.x];
    sAv[threadIdx.x] = Av[tok * (KVH_ * R_) + threadIdx.x];
  }
  __syncthreads();
  const float c  = cosb[(size_t)tok * (HD_ / 2) + (d & 63)];
  const float sn = sinb[(size_t)tok * (HD_ / 2) + (d & 63)];
  float rq[QR_];
#pragma unroll
  for (int r = 0; r < QR_; ++r) {
    float x  = Bq[(size_t)tok * (QR_ * HD_) + r * HD_ + d];
    float xp = Bq[(size_t)tok * (QR_ * HD_) + r * HD_ + (d ^ 64)];
    rq[r] = (d < 64) ? (x * c - xp * sn) : (x * c + xp * sn);
  }
  float rk[R_], bv[R_];
#pragma unroll
  for (int r = 0; r < R_; ++r) {
    float x  = Bk[(size_t)tok * (R_ * HD_) + r * HD_ + d];
    float xp = Bk[(size_t)tok * (R_ * HD_) + r * HD_ + (d ^ 64)];
    rk[r] = (d < 64) ? (x * c - xp * sn) : (x * c + xp * sn);
    bv[r] = Bv[(size_t)tok * (R_ * HD_) + r * HD_ + d];
  }
#pragma unroll
  for (int h = 0; h < H_; ++h) {
    float a = 0.f;
#pragma unroll
    for (int r = 0; r < QR_; ++r) a += sAq[h * QR_ + r] * rq[r];
    qo[(((size_t)(b * H_ + h)) * S_ + s) * HD_ + d] = f2bf(a * (1.0f / QR_));
  }
#pragma unroll
  for (int kv = 0; kv < KVH_; ++kv) {
    float ak = 0.f, av = 0.f;
#pragma unroll
    for (int r = 0; r < R_; ++r) {
      ak += sAk[kv * R_ + r] * rk[r];
      av += sAv[kv * R_ + r] * bv[r];
    }
    ko [(((size_t)(b * KVH_ + kv)) * S_ + s) * HD_ + d] = f2bf(ak * 0.5f);
    vto[(((size_t)(b * KVH_ + kv)) * HD_ + d) * S_ + s] = f2bf(av * 0.5f);
  }
}

// ---------------------------------------------------------------- flash attention
// Block = 4 waves sharing one (b,h) and a 64-query stripe. K tile (32x128) and
// V^T tile (128x32) are staged into LDS once per key block (TDM DMA when
// available, cooperative copy otherwise) and consumed by all 4 waves via
// ds loads feeding WMMA. Causal mask analytic; sink-gated softmax.
__global__ __launch_bounds__(128) void flash_attn_wmma(
    const unsigned short* __restrict__ qb,
    const unsigned short* __restrict__ kbm,
    const unsigned short* __restrict__ vt,
    const float* __restrict__ sinks,
    unsigned short* __restrict__ attn) {
  __shared__ __align__(16) unsigned short kbuf[32][HD_];    // 8 KB
  __shared__ __align__(16) unsigned short vbuf[HD_][32];    // 8 KB
  __shared__ __align__(16) unsigned short pbuf[4][16][32];  // 4 KB
  const int lane   = threadIdx.x & 31;
  const int widx   = threadIdx.x >> 5;
  const int lane16 = lane & 15;
  const int half   = lane >> 4;
  const int qblk = blockIdx.x & (S_ / 64 - 1);   // 32 stripes
  const int h    = (blockIdx.x >> 5) & (H_ - 1);
  const int b    = blockIdx.x >> 9;
  const int q0b  = qblk * 64;
  const int q0   = q0b + widx * 16;              // this wave's 16 queries
  const int kv   = h >> 2;                       // H/KVH = 4
  const unsigned short* qbase = qb  + ((size_t)(b * H_   + h ) * S_) * HD_;
  const unsigned short* kbase = kbm + ((size_t)(b * KVH_ + kv) * S_) * HD_;
  const unsigned short* vbase = vt  + ((size_t)(b * KVH_ + kv) * HD_) * S_;

  BF16Frag qf[4];
  {
    const unsigned short* qr = qbase + (size_t)(q0 + lane16) * HD_;
#pragma unroll
    for (int dsi = 0; dsi < 4; ++dsi) {
      qf[dsi].q[0] = *(const v4u*)(qr + dsi * 32 + 8 * half);
      qf[dsi].q[1] = *(const v4u*)(qr + dsi * 32 + 16 + 8 * half);
    }
  }
  float mrun[8], lrun[8];
  v8f acc[8];
  const v8f z8 = {};
#pragma unroll
  for (int i = 0; i < 8; ++i) { mrun[i] = -3.0e38f; lrun[i] = 0.f; }
#pragma unroll
  for (int dt = 0; dt < 8; ++dt) acc[dt] = z8;

  const int kend = q0b + 63;
  for (int kbs = 0; kbs <= kend; kbs += 32) {
    __syncthreads();   // previous iteration's LDS reads complete
#if HAVE_TDM
    if (widx == 0) {
      tdm_load_2d((unsigned int)(uintptr_t)&kbuf[0][0],
                  kbase + (size_t)kbs * HD_, HD_, 32, HD_);
      tdm_load_2d((unsigned int)(uintptr_t)&vbuf[0][0],
                  vbase + kbs, 32, HD_, S_);
      wait_tensorcnt0();
    }
#else
    {
      // cooperative stage: K rows kbs..kbs+31 (each thread: 32 elems)
      int row = threadIdx.x >> 2, part = (threadIdx.x & 3) * 32;
      const unsigned short* src = kbase + (size_t)(kbs + row) * HD_ + part;
#pragma unroll
      for (int j = 0; j < 4; ++j)
        *(v4u*)&kbuf[row][part + j * 8] = *(const v4u*)(src + j * 8);
      // V^T rows d=tid: 32 elems from column block kbs
      const unsigned short* vs = vbase + (size_t)threadIdx.x * S_ + kbs;
#pragma unroll
      for (int j = 0; j < 4; ++j)
        *(v4u*)&vbuf[threadIdx.x][j * 8] = *(const v4u*)(vs + j * 8);
    }
#endif
    __syncthreads();   // staged tiles visible to all waves

    v8f ct[2];
#pragma unroll
    for (int t = 0; t < 2; ++t) {
      v8f c = {};
      const unsigned short* kr = &kbuf[16 * t + lane16][0];
#pragma unroll
      for (int dsi = 0; dsi < 4; ++dsi) {
        BF16Frag bf;
        bf.q[0] = *(const v4u*)(kr + dsi * 32 + 8 * half);
        bf.q[1] = *(const v4u*)(kr + dsi * 32 + 16 + 8 * half);
        c = __builtin_amdgcn_wmma_f32_16x16x32_bf16(
            false, qf[dsi].v, false, bf.v, (short)0, c, false, false);
      }
      ct[t] = c;
    }
    // scale + causal mask
#pragma unroll
    for (int t = 0; t < 2; ++t) {
      int key = kbs + 16 * t + lane16;
#pragma unroll
      for (int i = 0; i < 8; ++i) {
        int row = q0 + i + 8 * half;
        float v = ct[t][i] * SCALING_;
        ct[t][i] = (key > row) ? NEG_ : v;
      }
    }
    float bm[8], so[8];
#pragma unroll
    for (int i = 0; i < 8; ++i) bm[i] = fmaxf(ct[0][i], ct[1][i]);
#pragma unroll
    for (int i = 0; i < 8; ++i)
#pragma unroll
      for (int m = 1; m < 16; m <<= 1) bm[i] = fmaxf(bm[i], __shfl_xor(bm[i], m, 32));
#pragma unroll
    for (int i = 0; i < 8; ++i) {
      float mn = fmaxf(mrun[i], bm[i]);
      so[i] = expf(mrun[i] - mn);
      mrun[i] = mn;
    }
#pragma unroll
    for (int i = 0; i < 8; ++i) {
      float p0 = expf(ct[0][i] - mrun[i]);
      float p1 = expf(ct[1][i] - mrun[i]);
      ct[0][i] = p0; ct[1][i] = p1;
      float s2 = p0 + p1;
#pragma unroll
      for (int m = 1; m < 16; m <<= 1) s2 += __shfl_xor(s2, m, 32);
      lrun[i] = lrun[i] * so[i] + s2;
    }
#pragma unroll
    for (int dt = 0; dt < 8; ++dt)
#pragma unroll
      for (int i = 0; i < 8; ++i) acc[dt][i] *= so[i];
    // transpose P (C layout -> A layout) via per-wave LDS slab
#pragma unroll
    for (int t = 0; t < 2; ++t)
#pragma unroll
      for (int i = 0; i < 8; ++i)
        pbuf[widx][i + 8 * half][16 * t + lane16] = f2bf(ct[t][i]);
    asm volatile("s_wait_dscnt 0" ::: "memory");
    BF16Frag pf;
    pf.q[0] = *(const v4u*)&pbuf[widx][lane16][8 * half];
    pf.q[1] = *(const v4u*)&pbuf[widx][lane16][16 + 8 * half];
#pragma unroll
    for (int dt = 0; dt < 8; ++dt) {
      BF16Frag bv;
      const unsigned short* vr = &vbuf[dt * 16 + lane16][8 * half];
      bv.q[0] = *(const v4u*)(vr);
      bv.q[1] = *(const v4u*)(vr + 16);
      acc[dt] = __builtin_amdgcn_wmma_f32_16x16x32_bf16(
          false, pf.v, false, bv.v, (short)0, acc[dt], false, false);
    }
  }
  const float sink = sinks[h];
  float fac[8];
#pragma unroll
  for (int i = 0; i < 8; ++i) {
    float lz = mrun[i] + logf(lrun[i]);               // logsumexp
    float gate = 1.0f / (1.0f + expf(-(lz - sink)));  // sigmoid
    fac[i] = gate / lrun[i];
  }
#pragma unroll
  for (int dt = 0; dt < 8; ++dt)
#pragma unroll
    for (int i = 0; i < 8; ++i) {
      int row = q0 + i + 8 * half;
      int col = h * HD_ + dt * 16 + lane16;
      attn[((size_t)(b * S_ + row)) * (H_ * HD_) + col] = f2bf(acc[dt][i] * fac[i]);
    }
}

// ---------------------------------------------------------------- host launcher
extern "C" void kernel_launch(void* const* d_in, const int* in_sizes, int n_in,
                              void* d_out, int out_size, void* d_ws, size_t ws_size,
                              hipStream_t stream) {
  (void)in_sizes; (void)n_in; (void)out_size; (void)ws_size;
  const float* hidden = (const float*)d_in[0];
  const float* cosb   = (const float*)d_in[1];
  const float* sinb   = (const float*)d_in[2];
  /* d_in[3] = attention_mask: causal, applied analytically */
  const float* W_A_q  = (const float*)d_in[4];
  const float* W_A_k  = (const float*)d_in[5];
  const float* W_A_v  = (const float*)d_in[6];
  const float* W_B_q  = (const float*)d_in[7];
  const float* W_B_k  = (const float*)d_in[8];
  const float* W_B_v  = (const float*)d_in[9];
  const float* W_o    = (const float*)d_in[10];
  const float* sinks  = (const float*)d_in[11];

  const int M = B_ * S_;                   // 4096 tokens
  const int NAq = H_ * QR_, NAk = KVH_ * R_, NBq = QR_ * HD_, NBk = R_ * HD_;

  char* ws = (char*)d_ws;
  size_t off = 0;
  auto alloc = [&](size_t bytes) -> char* {
    char* p = ws + off;
    off += (bytes + 255) & ~(size_t)255;
    return p;
  };
  unsigned short* hbf  = (unsigned short*)alloc((size_t)M * D_ * 2);
  unsigned short* wAq  = (unsigned short*)alloc((size_t)NAq * D_ * 2);
  unsigned short* wAk  = (unsigned short*)alloc((size_t)NAk * D_ * 2);
  unsigned short* wAv  = (unsigned short*)alloc((size_t)NAk * D_ * 2);
  unsigned short* wBq  = (unsigned short*)alloc((size_t)NBq * D_ * 2);
  unsigned short* wBk  = (unsigned short*)alloc((size_t)NBk * D_ * 2);
  unsigned short* wBv  = (unsigned short*)alloc((size_t)NBk * D_ * 2);
  unsigned short* wO   = (unsigned short*)alloc((size_t)D_ * D_ * 2);
  float* Aqf = (float*)alloc((size_t)M * NAq * 4);
  float* Akf = (float*)alloc((size_t)M * NAk * 4);
  float* Avf = (float*)alloc((size_t)M * NAk * 4);
  float* Bqf = (float*)alloc((size_t)M * NBq * 4);
  float* Bkf = (float*)alloc((size_t)M * NBk * 4);
  float* Bvf = (float*)alloc((size_t)M * NBk * 4);
  unsigned short* qbf  = (unsigned short*)alloc((size_t)B_ * H_ * S_ * HD_ * 2);
  unsigned short* kbf  = (unsigned short*)alloc((size_t)B_ * KVH_ * S_ * HD_ * 2);
  unsigned short* vtbf = (unsigned short*)alloc((size_t)B_ * KVH_ * HD_ * S_ * 2);
  unsigned short* attnbf = hbf;            // reuse: hidden-bf16 dead after projections

  // 1) hidden f32 -> bf16
  cvt_f32_bf16<<<(M * D_ + 255) / 256, 256, 0, stream>>>(hidden, hbf, M * D_);

  // 2) weights: transpose + convert (D x N -> N x D bf16)
  auto tr = [&](const float* src, unsigned short* dst, int N) {
    transpose_cvt<<<(D_ * N + 255) / 256, 256, 0, stream>>>(src, dst, D_, N);
  };
  tr(W_A_q, wAq, NAq); tr(W_A_k, wAk, NAk); tr(W_A_v, wAv, NAk);
  tr(W_B_q, wBq, NBq); tr(W_B_k, wBk, NBk); tr(W_B_v, wBv, NBk);
  tr(W_o,   wO,  D_);

  // 3) projection GEMMs (all K = D_)
  auto gemm = [&](const unsigned short* A, const unsigned short* Bt, float* C, int N) {
    int tiles = (M / 32) * ((N + 31) / 32);
    gemm_bf16_wmma<<<(tiles + 3) / 4, 128, 0, stream>>>(A, Bt, C, M, N, D_);
  };
  gemm(hbf, wAq, Aqf, NAq);
  gemm(hbf, wAk, Akf, NAk);
  gemm(hbf, wAv, Avf, NAk);
  gemm(hbf, wBq, Bqf, NBq);
  gemm(hbf, wBk, Bkf, NBk);
  gemm(hbf, wBv, Bvf, NBk);

  // 4) RoPE + rank combine -> q/k/vT bf16
  qkv_combine<<<M, 128, 0, stream>>>(Aqf, Akf, Avf, Bqf, Bkf, Bvf,
                                     cosb, sinb, qbf, kbf, vtbf);

  // 5) flash attention with sink gating -> attn (B,S,H*HD) bf16
  {
    int blocks = B_ * H_ * (S_ / 64);      // 1024 blocks, 4 waves each
    flash_attn_wmma<<<blocks, 128, 0, stream>>>(qbf, kbf, vtbf, sinks, attnbf);
  }

  // 6) output projection -> d_out f32 (B,S,D)
  {
    int tiles = (M / 32) * (D_ / 32);
    gemm_bf16_wmma<<<(tiles + 3) / 4, 128, 0, stream>>>(attnbf, wO, (float*)d_out,
                                                        M, D_, D_);
  }
}
